// PretrainVideoMamba_23673859736249
// MI455X (gfx1250) — matrix operand order repaired and checked by reading, not compiled, not run on hardware
//
#include <hip/hip_runtime.h>

typedef __bf16 bf16_t;
typedef __attribute__((ext_vector_type(16))) __bf16 v16bf;
typedef __attribute__((ext_vector_type(8)))  __bf16 v8bf;
typedef __attribute__((ext_vector_type(8)))  float  v8f;

constexpr int cE     = 384;
constexpr int cDI    = 768;
constexpr int cNS    = 16;
constexpr int cR     = 24;
constexpr int cDEPTH = 12;
constexpr int cL     = 1569;   // tokens
constexpr int cLP    = 1600;   // tokens padded to 32
constexpr int cF     = 8;
constexpr int cNP    = 196;    // patches per frame
constexpr int cDEC   = 512;
constexpr float cEPS = 1e-5f;

// ---------------------------------------------------------------------------
// bf16 WMMA GEMM body:  C[M,N] = A[M,K] * W[N,K]^T   (f32 accumulate)
// Each wave computes a 32x32 tile (2x2 WMMA tiles, fragments reused 2x).
// M % 32 == 0, N % 32 == 0, K % 32 == 0. 8 waves/block -> 32x256 of C.
// ---------------------------------------------------------------------------
__device__ __forceinline__ void gemm_tile32(
    const bf16_t* __restrict__ A, const bf16_t* __restrict__ W,
    float* __restrict__ C, int N, int K)
{
  const int lane = threadIdx.x & 31;
  const int wave = threadIdx.x >> 5;
  const int nt   = blockIdx.y * 8 + wave;    // 32-wide N tile index
  if (nt * 32 >= N) return;
  const int m0  = blockIdx.x * 32;
  const int n0  = nt * 32;
  const int row = lane & 15;
  const bool hi = lane >= 16;
  const int aoff = hi ? 8 : 0;
  const int boff = hi ? 8 : 0;

  const bf16_t* A0 = A + (size_t)(m0 + row) * K;
  const bf16_t* A1 = A0 + (size_t)16 * K;
  const bf16_t* W0 = W + (size_t)(n0 + row) * K;
  const bf16_t* W1 = W0 + (size_t)16 * K;

  v8f acc00 = {}, acc01 = {}, acc10 = {}, acc11 = {};
  for (int k0 = 0; k0 < K; k0 += 32) {
    union { v16bf v; v8bf h[2]; } a0, a1;
    a0.h[0] = *(const v8bf*)(A0 + k0 + aoff);
    a0.h[1] = *(const v8bf*)(A0 + k0 + aoff + 16);
    a1.h[0] = *(const v8bf*)(A1 + k0 + aoff);
    a1.h[1] = *(const v8bf*)(A1 + k0 + aoff + 16);
    union { v16bf v; v8bf h[2]; } bu0, bu1;
    bu0.h[0] = *(const v8bf*)(W0 + k0 + boff);
    bu0.h[1] = *(const v8bf*)(W0 + k0 + boff + 16);
    bu1.h[0] = *(const v8bf*)(W1 + k0 + boff);
    bu1.h[1] = *(const v8bf*)(W1 + k0 + boff + 16);
    v16bf b0 = bu0.v;
    v16bf b1 = bu1.v;
    acc00 = __builtin_amdgcn_wmma_f32_16x16x32_bf16(false, a0.v, false, b0, (short)0, acc00, false, false);
    acc01 = __builtin_amdgcn_wmma_f32_16x16x32_bf16(false, a0.v, false, b1, (short)0, acc01, false, false);
    acc10 = __builtin_amdgcn_wmma_f32_16x16x32_bf16(false, a1.v, false, b0, (short)0, acc10, false, false);
    acc11 = __builtin_amdgcn_wmma_f32_16x16x32_bf16(false, a1.v, false, b1, (short)0, acc11, false, false);
  }
  float* Cp = C + (size_t)(m0 + aoff + (hi ? 0 : 0) + (hi ? 8 : 0)) * N; // m0 + (hi?16:0); overwritten below
  Cp = C + (size_t)(m0 + (hi ? 8 : 0)) * N + n0 + row;
#pragma unroll
  for (int v = 0; v < 8; ++v) {
    Cp[(size_t)v * N]           = acc00[v];
    Cp[(size_t)v * N + 16]      = acc01[v];
    Cp[(size_t)(v + 16) * N]    = acc10[v];
    Cp[(size_t)(v + 16) * N + 16] = acc11[v];
  }
}

__global__ __launch_bounds__(256) void k_gemm_bf16(
    const bf16_t* __restrict__ A, const bf16_t* __restrict__ W,
    float* __restrict__ C, int N, int K)
{
  gemm_tile32(A, W, C, N, K);
}

// batched over 2 directions via blockIdx.z
__global__ __launch_bounds__(256) void k_gemm_bf16_b2(
    const bf16_t* __restrict__ A, size_t strA,
    const bf16_t* __restrict__ Wq0, const bf16_t* __restrict__ Wq1,
    float* __restrict__ C, size_t strC, int N, int K)
{
  const int dir = blockIdx.z;
  gemm_tile32(A + (size_t)dir * strA, dir ? Wq1 : Wq0, C + (size_t)dir * strC, N, K);
}

// ---------------------------------------------------------------------------
// weight conversion / padding kernels
// ---------------------------------------------------------------------------
__global__ void k_convert(const float* __restrict__ s, bf16_t* __restrict__ d, int n) {
  int i = blockIdx.x * blockDim.x + threadIdx.x;
  if (i < n) d[i] = (bf16_t)s[i];
}
// x_proj (56 x 768) -> padded (64 x 768), zeros for n>=56, per layer
__global__ void k_pad_xpw(const float* __restrict__ s, bf16_t* __restrict__ d, int total) {
  int i = blockIdx.x * blockDim.x + threadIdx.x;
  if (i >= total) return;
  int k = i % cDI; int n = (i / cDI) % 64; int l = i / (cDI * 64);
  d[i] = (n < 56) ? (bf16_t)s[((size_t)l * 56 + n) * cDI + k] : (bf16_t)0.0f;
}
// dt_w (768 x 24) -> padded (768 x 32), zeros for k>=24, per layer
__global__ void k_pad_dtw(const float* __restrict__ s, bf16_t* __restrict__ d, int total) {
  int i = blockIdx.x * blockDim.x + threadIdx.x;
  if (i >= total) return;
  int k = i % 32; int dd = (i / 32) % cDI; int l = i / (32 * cDI);
  d[i] = (k < cR) ? (bf16_t)s[((size_t)l * cDI + dd) * cR + k] : (bf16_t)0.0f;
}

// ---------------------------------------------------------------------------
// patch gather: A[row=f*196+py*14+px][c*256+iy*16+ix] = x[0,c,f,py*16+iy,px*16+ix]
// ---------------------------------------------------------------------------
__global__ void k_patch_gather(const float* __restrict__ x, bf16_t* __restrict__ A) {
  int i = blockIdx.x * blockDim.x + threadIdx.x;
  if (i >= 1568 * 768) return;
  int col = i % 768, row = i / 768;
  int f = row / cNP, p = row % cNP, py = p / 14, px = p % 14;
  int c = col / 256, r = col % 256, iy = r / 16, ix = r % 16;
  size_t src = ((size_t)(c * cF + f) * 224 + (py * 16 + iy)) * 224 + (px * 16 + ix);
  A[i] = (bf16_t)x[src];
}

// ---------------------------------------------------------------------------
// assemble hid tokens: h=hid, res=0
// ---------------------------------------------------------------------------
__global__ void k_assemble_hid(const float* __restrict__ feat, const float* __restrict__ pb,
                               const float* __restrict__ cls, const float* __restrict__ pe,
                               const float* __restrict__ tp, float* __restrict__ h,
                               float* __restrict__ res) {
  int i = blockIdx.x * blockDim.x + threadIdx.x;
  if (i >= cLP * cE) return;
  int e = i % cE, t = i / cE;
  float v = 0.0f;
  if (t == 0) v = cls[e] + pe[e];
  else if (t < cL) {
    int r = t - 1, f = r / cNP, p = r % cNP;
    v = feat[(size_t)r * cE + e] + pb[e] + pe[(size_t)(1 + p) * cE + e] + tp[(size_t)f * cE + e];
  }
  h[i] = v; res[i] = 0.0f;
}

// ---------------------------------------------------------------------------
// res += h ; hn_bf16 = bf16(rmsnorm(res) * nw). Pad rows -> zero.
// ---------------------------------------------------------------------------
__global__ __launch_bounds__(384) void k_add_rms(
    float* __restrict__ res, const float* __restrict__ h,
    const float* __restrict__ nw, bf16_t* __restrict__ hn)
{
  __shared__ float red[512];
  int t = blockIdx.x, e = threadIdx.x;
  if (t >= cL) { hn[(size_t)t * cE + e] = (bf16_t)0.0f; return; }
  size_t idx = (size_t)t * cE + e;
  float v = res[idx] + h[idx];
  res[idx] = v;
  red[e] = v * v;
  if (e < 128) red[384 + e] = 0.0f;
  __syncthreads();
  for (int s = 256; s > 0; s >>= 1) { if (e < s) red[e] += red[e + s]; __syncthreads(); }
  float inv = rsqrtf(red[0] / (float)cE + cEPS);
  hn[idx] = (bf16_t)(v * inv * nw[e]);
}

// ---------------------------------------------------------------------------
// causal depthwise conv K=4 + silu, both directions (blockIdx.y = dir).
// dir=1 operates on the reversed sequence. Pad rows -> zero.
// ---------------------------------------------------------------------------
__global__ void k_conv_silu(const float* __restrict__ xz,
                            const float* __restrict__ cwf, const float* __restrict__ cwb,
                            const float* __restrict__ cbf, const float* __restrict__ cbb,
                            float* __restrict__ xc, bf16_t* __restrict__ xcb)
{
  const int dir = blockIdx.y;
  int i = blockIdx.x * blockDim.x + threadIdx.x;
  if (i >= cLP * cDI) return;
  const float* cw = dir ? cwb : cwf;
  const float* cb = dir ? cbb : cbf;
  float*  out  = xc  + (size_t)dir * cLP * cDI;
  bf16_t* outb = xcb + (size_t)dir * cLP * cDI;
  int d = i % cDI, t = i / cDI;
  if (t >= cL) { out[i] = 0.0f; outb[i] = (bf16_t)0.0f; return; }
  float acc = cb[d];
  float w0 = cw[d * 4 + 0], w1 = cw[d * 4 + 1], w2 = cw[d * 4 + 2], w3 = cw[d * 4 + 3];
  int tm;
  tm = t - 3; if (tm >= 0) acc += w0 * xz[(size_t)(dir ? cL - 1 - tm : tm) * 1536 + d];
  tm = t - 2; if (tm >= 0) acc += w1 * xz[(size_t)(dir ? cL - 1 - tm : tm) * 1536 + d];
  tm = t - 1; if (tm >= 0) acc += w2 * xz[(size_t)(dir ? cL - 1 - tm : tm) * 1536 + d];
  acc += w3 * xz[(size_t)(dir ? cL - 1 - t : t) * 1536 + d];
  float v = acc / (1.0f + __expf(-acc));
  out[i] = v; outb[i] = (bf16_t)v;
}

// xdbl[:, :24] -> bf16, padded to 32 cols with zeros, both dirs (blockIdx.y)
__global__ void k_dtin_pad(const float* __restrict__ xdbl, bf16_t* __restrict__ d) {
  const int dir = blockIdx.y;
  int i = blockIdx.x * blockDim.x + threadIdx.x;
  if (i >= cLP * 32) return;
  int k = i % 32, t = i / 32;
  d[(size_t)dir * cLP * 32 + i] =
      (k < cR) ? (bf16_t)xdbl[(size_t)dir * cLP * 64 + (size_t)t * 64 + k] : (bf16_t)0.0f;
}

// ---------------------------------------------------------------------------
// selective-scan, BOTH directions in one launch (blocks 0..2 fwd, 3..5 bwd).
// One thread per channel, 16 states in registers, B/C staged via LDS.
// dir=1 scatters its output to the original token index.
// ---------------------------------------------------------------------------
__global__ __launch_bounds__(256) void k_ssm_scan(
    const float* __restrict__ xc, const float* __restrict__ xdbl,
    const float* __restrict__ dtm, const float* __restrict__ xz,
    const float* __restrict__ Alog0, const float* __restrict__ Alog1,
    const float* __restrict__ Dv0, const float* __restrict__ Dv1,
    const float* __restrict__ dtb0, const float* __restrict__ dtb1,
    float* __restrict__ yout)
{
  __shared__ float sbc[64 * 32];
  const int dir = (blockIdx.x >= 3) ? 1 : 0;
  const int d = (blockIdx.x - dir * 3) * 256 + threadIdx.x;   // 0..767
  const float* Alog = dir ? Alog1 : Alog0;
  const float* Dv   = dir ? Dv1   : Dv0;
  const float* dtbp = dir ? dtb1  : dtb0;
  xc   += (size_t)dir * cLP * cDI;
  xdbl += (size_t)dir * cLP * 64;
  dtm  += (size_t)dir * cLP * cDI;
  yout += (size_t)dir * cLP * cDI;

  float A[16], h[16];
#pragma unroll
  for (int n = 0; n < 16; ++n) { A[n] = -__expf(Alog[(size_t)d * cNS + n]); h[n] = 0.0f; }
  float Dd = Dv[d], dtbd = dtbp[d];

  for (int t0 = 0; t0 < cL; t0 += 64) {
    int nt = min(64, cL - t0);
    __syncthreads();
    for (int i = threadIdx.x; i < nt * 32; i += 256) {
      int tt = i >> 5, c = i & 31;
      sbc[i] = xdbl[(size_t)(t0 + tt) * 64 + 24 + c];
    }
    __syncthreads();
    for (int tt = 0; tt < nt; ++tt) {
      int t = t0 + tt;
      float xdt = dtm[(size_t)t * cDI + d] + dtbd;
      float dt = (xdt > 20.0f) ? xdt : log1pf(__expf(xdt));     // softplus
      float u = xc[(size_t)t * cDI + d];
      float du = dt * u;
      const float* Bt = &sbc[tt * 32];
      const float* Ct = &sbc[tt * 32 + 16];
      float acc = 0.0f;
#pragma unroll
      for (int n = 0; n < 16; ++n) {
        h[n] = __expf(dt * A[n]) * h[n] + du * Bt[n];
        acc += h[n] * Ct[n];
      }
      int ot = dir ? (cL - 1 - t) : t;
      float zl = xz[(size_t)ot * 1536 + cDI + d];
      float y = (acc + u * Dd) * (zl / (1.0f + __expf(-zl)));
      yout[(size_t)ot * cDI + d] = y;
    }
  }
}

// ybf = bf16(y_fwd + y_bwd); pad rows -> zero (clean WMMA operand)
__global__ void k_ymerge(const float* __restrict__ y, bf16_t* __restrict__ ybf) {
  int i = blockIdx.x * blockDim.x + threadIdx.x;
  if (i >= cLP * cDI) return;
  int t = i / cDI;
  ybf[i] = (t < cL) ? (bf16_t)(y[i] + y[(size_t)cLP * cDI + i]) : (bf16_t)0.0f;
}

// ---------------------------------------------------------------------------
// xf = rmsnorm(h+res)*nfw ; xfc_bf = bf16(xf + clip_pos). Pad rows -> zero.
// ---------------------------------------------------------------------------
__global__ __launch_bounds__(384) void k_final_rms(
    const float* __restrict__ h, const float* __restrict__ res,
    const float* __restrict__ nfw, const float* __restrict__ cpos,
    float* __restrict__ xf, bf16_t* __restrict__ xfc)
{
  __shared__ float red[512];
  int t = blockIdx.x, e = threadIdx.x;
  size_t idx = (size_t)t * cE + e;
  if (t >= cL) { xf[idx] = 0.0f; xfc[idx] = (bf16_t)0.0f; return; }
  float v = res[idx] + h[idx];
  red[e] = v * v;
  if (e < 128) red[384 + e] = 0.0f;
  __syncthreads();
  for (int s = 256; s > 0; s >>= 1) { if (e < s) red[e] += red[e + s]; __syncthreads(); }
  float inv = rsqrtf(red[0] / (float)cE + cEPS);
  float o = v * inv * nfw[e];
  xf[idx] = o;
  xfc[idx] = (bf16_t)(o + cpos[idx]);
}

// layernorm over 512 of (xclip + dec_b) -> out
__global__ __launch_bounds__(512) void k_clip_ln(
    const float* __restrict__ xc, const float* __restrict__ db,
    const float* __restrict__ lw, const float* __restrict__ lb,
    float* __restrict__ out)
{
  __shared__ float red[512];
  int t = blockIdx.x, o = threadIdx.x;
  float v = xc[(size_t)t * cDEC + o] + db[o];
  red[o] = v; __syncthreads();
  for (int s = 256; s > 0; s >>= 1) { if (o < s) red[o] += red[o + s]; __syncthreads(); }
  float m = red[0] / (float)cDEC;
  __syncthreads();
  red[o] = (v - m) * (v - m); __syncthreads();
  for (int s = 256; s > 0; s >>= 1) { if (o < s) red[o] += red[o + s]; __syncthreads(); }
  float inv = rsqrtf(red[0] / (float)cDEC + cEPS);
  out[(size_t)t * cDEC + o] = (v - m) * inv * lw[o] + lb[o];
}

// x_pool = layernorm(xf[0] + mean(xf[1:]))
__global__ __launch_bounds__(384) void k_pool(
    const float* __restrict__ xf, const float* __restrict__ pw,
    const float* __restrict__ pb, float* __restrict__ out)
{
  __shared__ float red[512];
  int e = threadIdx.x;
  float s = 0.0f;
  for (int t = 1; t < cL; ++t) s += xf[(size_t)t * cE + e];
  float v = xf[e] + s / (float)(cL - 1);
  red[e] = v;
  if (e < 128) red[384 + e] = 0.0f;
  __syncthreads();
  for (int st = 256; st > 0; st >>= 1) { if (e < st) red[e] += red[e + st]; __syncthreads(); }
  float m = red[0] / (float)cE;
  __syncthreads();
  red[e] = (v - m) * (v - m);
  if (e < 128) red[384 + e] = 0.0f;
  __syncthreads();
  for (int st = 256; st > 0; st >>= 1) { if (e < st) red[e] += red[e + st]; __syncthreads(); }
  float inv = rsqrtf(red[0] / (float)cE + cEPS);
  out[e] = (v - m) * inv * pw[e] + pb[e];
}

// ---------------------------------------------------------------------------
extern "C" void kernel_launch(void* const* d_in, const int* in_sizes, int n_in,
                              void* d_out, int out_size, void* d_ws, size_t ws_size,
                              hipStream_t stream)
{
  (void)in_sizes; (void)n_in; (void)out_size; (void)ws_size;
  const float* x        = (const float*)d_in[0];
  const float* patch_w  = (const float*)d_in[1];
  const float* patch_b  = (const float*)d_in[2];
  const float* cls_tok  = (const float*)d_in[3];
  const float* pos_emb  = (const float*)d_in[4];
  const float* temp_pos = (const float*)d_in[5];
  const float* norm_w   = (const float*)d_in[6];
  const float* in_proj  = (const float*)d_in[7];
  const float* conv_w[2] = {(const float*)d_in[8],  (const float*)d_in[15]};
  const float* conv_b[2] = {(const float*)d_in[9],  (const float*)d_in[16]};
  const float* xpw[2]    = {(const float*)d_in[10], (const float*)d_in[17]};
  const float* dtw[2]    = {(const float*)d_in[11], (const float*)d_in[18]};
  const float* dtb[2]    = {(const float*)d_in[12], (const float*)d_in[19]};
  const float* alog[2]   = {(const float*)d_in[13], (const float*)d_in[20]};
  const float* Dp[2]     = {(const float*)d_in[14], (const float*)d_in[21]};
  const float* out_proj = (const float*)d_in[22];
  const float* norm_f_w = (const float*)d_in[23];
  const float* pool_w   = (const float*)d_in[24];
  const float* pool_b   = (const float*)d_in[25];
  const float* clip_pos = (const float*)d_in[26];
  const float* dec_w    = (const float*)d_in[27];
  const float* dec_b    = (const float*)d_in[28];
  const float* dec_ln_w = (const float*)d_in[29];
  const float* dec_ln_b = (const float*)d_in[30];

  char* base = (char*)d_ws;
  size_t off = 0;
  auto ALLOC = [&](size_t nbytes) -> char* {
    char* p = base + off;
    off = (off + nbytes + 255) & ~(size_t)255;
    return p;
  };
  bf16_t* w_ip  = (bf16_t*)ALLOC((size_t)cDEPTH * 1536 * cE * 2);
  bf16_t* w_xp  = (bf16_t*)ALLOC((size_t)2 * cDEPTH * 64 * cDI * 2);
  bf16_t* w_dt  = (bf16_t*)ALLOC((size_t)2 * cDEPTH * cDI * 32 * 2);
  bf16_t* w_op  = (bf16_t*)ALLOC((size_t)cDEPTH * cE * cDI * 2);
  bf16_t* w_pw  = (bf16_t*)ALLOC((size_t)cE * 768 * 2);
  bf16_t* w_dec = (bf16_t*)ALLOC((size_t)cDEC * cE * 2);
  bf16_t* apat  = (bf16_t*)ALLOC((size_t)1568 * 768 * 2);
  float*  featb = (float*)ALLOC((size_t)1568 * cE * 4);
  float*  hbuf  = (float*)ALLOC((size_t)cLP * cE * 4);
  float*  resb  = (float*)ALLOC((size_t)cLP * cE * 4);
  bf16_t* hn_bf = (bf16_t*)ALLOC((size_t)cLP * cE * 2);
  float*  xzb   = (float*)ALLOC((size_t)cLP * 1536 * 4);
  float*  xcf   = (float*)ALLOC((size_t)2 * cLP * cDI * 4);
  bf16_t* xcb   = (bf16_t*)ALLOC((size_t)2 * cLP * cDI * 2);
  float*  xdb   = (float*)ALLOC((size_t)2 * cLP * 64 * 4);
  bf16_t* dti   = (bf16_t*)ALLOC((size_t)2 * cLP * 32 * 2);
  float*  dtf   = (float*)ALLOC((size_t)2 * cLP * cDI * 4);
  float*  yrw   = (float*)ALLOC((size_t)2 * cLP * cDI * 4);
  bf16_t* ybf   = (bf16_t*)ALLOC((size_t)cLP * cDI * 2);
  float*  xfb   = (float*)ALLOC((size_t)cLP * cE * 4);
  bf16_t* xfc   = (bf16_t*)ALLOC((size_t)cLP * cE * 2);
  float*  xclip = (float*)ALLOC((size_t)cLP * cDEC * 4);

  auto g1 = [](size_t n, int b) { return dim3((unsigned)((n + (size_t)b - 1) / b)); };

  // ---- weight conversion (bf16) ----
  { int n = cDEPTH * 1536 * cE; k_convert<<<g1(n,256),256,0,stream>>>(in_proj, w_ip, n); }
  { int n = cDEPTH * 64 * cDI;
    k_pad_xpw<<<g1(n,256),256,0,stream>>>(xpw[0], w_xp, n);
    k_pad_xpw<<<g1(n,256),256,0,stream>>>(xpw[1], w_xp + (size_t)cDEPTH*64*cDI, n); }
  { int n = cDEPTH * cDI * 32;
    k_pad_dtw<<<g1(n,256),256,0,stream>>>(dtw[0], w_dt, n);
    k_pad_dtw<<<g1(n,256),256,0,stream>>>(dtw[1], w_dt + (size_t)cDEPTH*cDI*32, n); }
  { int n = cDEPTH * cE * cDI; k_convert<<<g1(n,256),256,0,stream>>>(out_proj, w_op, n); }
  { int n = cE * 768;  k_convert<<<g1(n,256),256,0,stream>>>(patch_w, w_pw, n); }
  { int n = cDEC * cE; k_convert<<<g1(n,256),256,0,stream>>>(dec_w, w_dec, n); }

  // ---- patch embedding ----
  k_patch_gather<<<g1((size_t)1568*768,256),256,0,stream>>>(x, apat);
  k_gemm_bf16<<<dim3(1568/32, 2), 256, 0, stream>>>(apat, w_pw, featb, cE, 768);
  k_assemble_hid<<<g1((size_t)cLP*cE,256),256,0,stream>>>(featb, patch_b, cls_tok,
                                                          pos_emb, temp_pos, hbuf, resb);

  // ---- 12 Mamba blocks ----
  const int MT = cLP / 32;  // 50
  for (int l = 0; l < cDEPTH; ++l) {
    k_add_rms<<<cLP, 384, 0, stream>>>(resb, hbuf, norm_w + (size_t)l * cE, hn_bf);
    k_gemm_bf16<<<dim3(MT, 6), 256, 0, stream>>>(hn_bf, w_ip + (size_t)l*1536*cE,
                                                 xzb, 1536, cE);
    k_conv_silu<<<dim3((unsigned)(((size_t)cLP*cDI + 255)/256), 2), 256, 0, stream>>>(
        xzb, conv_w[0] + (size_t)l*cDI*4, conv_w[1] + (size_t)l*cDI*4,
        conv_b[0] + (size_t)l*cDI, conv_b[1] + (size_t)l*cDI, xcf, xcb);
    k_gemm_bf16_b2<<<dim3(MT, 1, 2), 256, 0, stream>>>(
        xcb, (size_t)cLP*cDI,
        w_xp + (size_t)l*64*cDI, w_xp + ((size_t)cDEPTH + l)*64*cDI,
        xdb, (size_t)cLP*64, 64, cDI);
    k_dtin_pad<<<dim3((unsigned)(((size_t)cLP*32 + 255)/256), 2), 256, 0, stream>>>(xdb, dti);
    k_gemm_bf16_b2<<<dim3(MT, 3, 2), 256, 0, stream>>>(
        dti, (size_t)cLP*32,
        w_dt + (size_t)l*cDI*32, w_dt + ((size_t)cDEPTH + l)*cDI*32,
        dtf, (size_t)cLP*cDI, cDI, 32);
    k_ssm_scan<<<6, 256, 0, stream>>>(
        xcf, xdb, dtf, xzb,
        alog[0] + (size_t)l*cDI*cNS, alog[1] + (size_t)l*cDI*cNS,
        Dp[0] + (size_t)l*cDI, Dp[1] + (size_t)l*cDI,
        dtb[0] + (size_t)l*cDI, dtb[1] + (size_t)l*cDI, yrw);
    k_ymerge<<<g1((size_t)cLP*cDI,256),256,0,stream>>>(yrw, ybf);
    k_gemm_bf16<<<dim3(MT, 2), 256, 0, stream>>>(ybf, w_op + (size_t)l*cE*cDI,
                                                 hbuf, cE, cDI);
  }

  // ---- head ----
  k_final_rms<<<cLP, 384, 0, stream>>>(hbuf, resb, norm_f_w, clip_pos, xfb, xfc);
  k_gemm_bf16<<<dim3(MT, 2), 256, 0, stream>>>(xfc, w_dec, xclip, cDEC, cE);
  k_clip_ln<<<cL, 512, 0, stream>>>(xclip, dec_b, dec_ln_w, dec_ln_b, (float*)d_out);
  k_pool<<<1, 384, 0, stream>>>(xfb, pool_w, pool_b, (float*)d_out + (size_t)cL * cDEC);
}
